// Fp8Linear_59846074302839
// MI455X (gfx1250) — 8-bit matrix operand order repaired and checked by reading, compiled, not run on hardware
//
#include <hip/hip_runtime.h>


typedef __attribute__((ext_vector_type(16))) int   v16i;
typedef __attribute__((ext_vector_type(8)))  float v8f;

typedef int v4i_b128 __attribute__((__vector_size__(16)));
typedef __attribute__((address_space(1))) v4i_b128* as1_v4i_ptr;
typedef __attribute__((address_space(3))) v4i_b128* as3_v4i_ptr;

#define F8_MAX 448.0f
#define EPSQ   1e-12f

#if defined(__has_builtin)
#if __has_builtin(__builtin_amdgcn_cvt_pk_fp8_f32)
#define HAVE_CVT_FP8 1
#endif
#if __has_builtin(__builtin_amdgcn_global_load_async_to_lds_b128) && \
    __has_builtin(__builtin_amdgcn_s_wait_asynccnt)
#define HAVE_ASYNC_LDS 1
#endif
#endif

__device__ __forceinline__ unsigned int f32_to_e4m3_byte(float f) {
    float cf = fminf(fmaxf(f, -F8_MAX), F8_MAX);
    unsigned int u    = __float_as_uint(cf);
    unsigned int sign = (u >> 24) & 0x80u;
    unsigned int ab   = u & 0x7FFFFFFFu;
    unsigned int r;
    if (ab >= 0x3C800000u) {
        unsigned int m = ab + 0x0007FFFFu + ((ab >> 20) & 1u);
        int e = (int)((m >> 23) & 0xFFu) - 127;
        r = (unsigned int)(((e + 7) << 3) | ((m >> 20) & 7u));
    } else {
        float a = __uint_as_float(ab);
        int   m = (int)rintf(a * 512.0f);
        r = (unsigned int)m;
    }
    return r | sign;
}

__device__ __forceinline__ unsigned int pack4_fp8(float a, float b, float c, float d) {
#ifdef HAVE_CVT_FP8
    int r = 0;
    r = __builtin_amdgcn_cvt_pk_fp8_f32(a, b, r, false);
    r = __builtin_amdgcn_cvt_pk_fp8_f32(c, d, r, true);
    return (unsigned int)r;
#else
    return f32_to_e4m3_byte(a) | (f32_to_e4m3_byte(b) << 8) |
           (f32_to_e4m3_byte(c) << 16) | (f32_to_e4m3_byte(d) << 24);
#endif
}

__device__ __forceinline__ void copy16_g2l(const unsigned char* __restrict__ g,
                                           unsigned char* l) {
#ifdef HAVE_ASYNC_LDS
    __builtin_amdgcn_global_load_async_to_lds_b128(
        (as1_v4i_ptr)(void*)g,
        (as3_v4i_ptr)(void*)l,
         0,  0);
#else
    *(uint4*)l = *(const uint4*)g;
#endif
}

__device__ __forceinline__ void g2l_wait_all() {
#ifdef HAVE_ASYNC_LDS
    __builtin_amdgcn_s_wait_asynccnt(0);
#endif
}

__global__ void k_init_amax(unsigned int* amax_bits) {
    if (threadIdx.x == 0 && blockIdx.x == 0) *amax_bits = 0u;
}

__global__ __launch_bounds__(256) void k_amax(const float* __restrict__ x,
                                              unsigned int* __restrict__ amax_bits,
                                              long n4) {
    const float4* x4 = (const float4*)x;
    long i      = (long)blockIdx.x * blockDim.x + threadIdx.x;
    long stride = (long)gridDim.x * blockDim.x;
    float m = 0.0f;
    for (; i < n4; i += stride) {
        float4 v = x4[i];
        m = fmaxf(m, fmaxf(fmaxf(fabsf(v.x), fabsf(v.y)),
                           fmaxf(fabsf(v.z), fabsf(v.w))));
    }
    __shared__ float red[256];
    red[threadIdx.x] = m;
    __syncthreads();
    for (int s = 128; s > 0; s >>= 1) {
        if ((int)threadIdx.x < s)
            red[threadIdx.x] = fmaxf(red[threadIdx.x], red[threadIdx.x + s]);
        __syncthreads();
    }
    if (threadIdx.x == 0) atomicMax(amax_bits, __float_as_uint(red[0]));
}

__global__ __launch_bounds__(256) void k_quant_x(const float* __restrict__ x,
                                                 const float* __restrict__ amax,
                                                 unsigned int* __restrict__ q,
                                                 long n4) {
    long i = (long)blockIdx.x * blockDim.x + threadIdx.x;
    if (i >= n4) return;
    float s = F8_MAX / fmaxf(*amax, EPSQ);
    float4 v = ((const float4*)x)[i];
    float a = fminf(fmaxf(v.x * s, -F8_MAX), F8_MAX);
    float b = fminf(fmaxf(v.y * s, -F8_MAX), F8_MAX);
    float c = fminf(fmaxf(v.z * s, -F8_MAX), F8_MAX);
    float d = fminf(fmaxf(v.w * s, -F8_MAX), F8_MAX);
    q[i] = pack4_fp8(a, b, c, d);
}

__global__ __launch_bounds__(256) void k_quant_w(const float* __restrict__ w,
                                                 unsigned int* __restrict__ q,
                                                 long n4) {
    long i = (long)blockIdx.x * blockDim.x + threadIdx.x;
    if (i >= n4) return;
    float4 v = ((const float4*)w)[i];
    q[i] = pack4_fp8(v.x, v.y, v.z, v.w);
}

#define BM 128
#define BN 128
#define BK 128
#define LDA 144

__device__ __forceinline__ void issue_tile(const unsigned char* __restrict__ gA,
                                           const unsigned char* __restrict__ gB,
                                           unsigned char* lA, unsigned char* lB,
                                           int tid, long Kl) {
    #pragma unroll
    for (int i = 0; i < 4; ++i) {
        int linear = tid + i * 256;
        int row = linear >> 3;
        int col = (linear & 7) << 4;
        copy16_g2l(gA + (long)row * Kl + col, lA + row * LDA + col);
        copy16_g2l(gB + (long)row * Kl + col, lB + row * LDA + col);
    }
}

__global__ __launch_bounds__(256) void k_gemm_fp8(
    const unsigned char* __restrict__ qx,
    const unsigned char* __restrict__ qw,
    const float* __restrict__ amax,
    const float* __restrict__ w_scale,
    const float* __restrict__ bias,
    float* __restrict__ out,
    int M, int N, int K)
{
    __shared__ __align__(16) unsigned char As[2][BM * LDA];
    __shared__ __align__(16) unsigned char Bs[2][BN * LDA];

    const int tid  = threadIdx.x;
    const int lane = tid & 31;
    const int wave = tid >> 5;
    const int wm   = wave & 3;
    const int wn   = wave >> 2;
    const int ln   = lane & 15;
    const int lh   = lane >> 4;

    const int bm0 = blockIdx.y * BM;
    const int bn0 = blockIdx.x * BN;
    const long Kl = (long)K;

    v8f acc[2][4];
    {
        v8f z = {};
        #pragma unroll
        for (int i = 0; i < 2; ++i)
            #pragma unroll
            for (int j = 0; j < 4; ++j) acc[i][j] = z;
    }

    const unsigned char* gA0 = qx + (long)bm0 * Kl;
    const unsigned char* gB0 = qw + (long)bn0 * Kl;

    const int kTiles = K / BK;

    issue_tile(gA0, gB0, &As[0][0], &Bs[0][0], tid, Kl);

    for (int kt = 0; kt < kTiles; ++kt) {
        const int cur = kt & 1;

        g2l_wait_all();
        __syncthreads();

        if (kt + 1 < kTiles)
            issue_tile(gA0 + (long)(kt + 1) * BK, gB0 + (long)(kt + 1) * BK,
                       &As[cur ^ 1][0], &Bs[cur ^ 1][0], tid, Kl);

        v16i bfrag[4];
        #pragma unroll
        for (int tn = 0; tn < 4; ++tn) {
            const unsigned char* p = &Bs[cur][(wn * 64 + tn * 16 + ln) * LDA];
            uint4 q0 = *(const uint4*)(p + (0 * 32 + lh * 16));
            uint4 q1 = *(const uint4*)(p + (1 * 32 + lh * 16));
            uint4 q2 = *(const uint4*)(p + (2 * 32 + lh * 16));
            uint4 q3 = *(const uint4*)(p + (3 * 32 + lh * 16));
            v16i bf;
            bf[0]  = (int)q0.x; bf[1]  = (int)q0.y; bf[2]  = (int)q0.z; bf[3]  = (int)q0.w;
            bf[4]  = (int)q1.x; bf[5]  = (int)q1.y; bf[6]  = (int)q1.z; bf[7]  = (int)q1.w;
            bf[8]  = (int)q2.x; bf[9]  = (int)q2.y; bf[10] = (int)q2.z; bf[11] = (int)q2.w;
            bf[12] = (int)q3.x; bf[13] = (int)q3.y; bf[14] = (int)q3.z; bf[15] = (int)q3.w;
            bfrag[tn] = bf;
        }

        #pragma unroll
        for (int tm = 0; tm < 2; ++tm) {
            const unsigned char* p = &As[cur][(wm * 32 + tm * 16 + ln) * LDA];
            v16i af;
            #pragma unroll
            for (int j = 0; j < 8; ++j) {
                uint2 d = *(const uint2*)(p + (j >> 1) * 32 + lh * 16 + (j & 1) * 8);
                af[2 * j]     = (int)d.x;
                af[2 * j + 1] = (int)d.y;
            }
            #pragma unroll
            for (int tn = 0; tn < 4; ++tn) {
                acc[tm][tn] = __builtin_amdgcn_wmma_f32_16x16x128_fp8_fp8(
                    af, bfrag[tn], (short)0, acc[tm][tn],
                     false,  false);
            }
        }
    }

    const float scale = (fmaxf(*amax, EPSQ) / F8_MAX) * (*w_scale);
    #pragma unroll
    for (int tm = 0; tm < 2; ++tm) {
        #pragma unroll
        for (int tn = 0; tn < 4; ++tn) {
            const int n  = bn0 + wn * 64 + tn * 16 + ln;
            const float bsv = bias[n];
            #pragma unroll
            for (int r = 0; r < 8; ++r) {
                const int m = bm0 + wm * 32 + tm * 16 + r + 8 * lh;
                __builtin_nontemporal_store(acc[tm][tn][r] * scale + bsv,
                                            &out[(long)m * N + n]);
            }
        }
    }
}

extern "C" void kernel_launch(void* const* d_in, const int* in_sizes, int n_in,
                              void* d_out, int out_size, void* d_ws, size_t ws_size,
                              hipStream_t stream) {
    const float* x       = (const float*)d_in[0];
    const float* qweight = (const float*)d_in[1];
    const float* w_scale = (const float*)d_in[2];
    const float* bias    = (const float*)d_in[3];
    float*       out     = (float*)d_out;

    const int  N  = in_sizes[3];
    const long Kl = (long)in_sizes[1] / N;
    const int  K  = (int)Kl;
    const int  M  = (int)((long)in_sizes[0] / Kl);

    unsigned char* ws        = (unsigned char*)d_ws;
    unsigned int*  amax_bits = (unsigned int*)ws;
    float*         amax      = (float*)ws;
    unsigned char* qx        = ws + 256;
    unsigned char* qw        = qx + (long)M * K;

    k_init_amax<<<1, 1, 0, stream>>>(amax_bits);

    const long n4x = (long)M * K / 4;
    k_amax<<<2048, 256, 0, stream>>>(x, amax_bits, n4x);

    k_quant_x<<<(unsigned)((n4x + 255) / 256), 256, 0, stream>>>(
        x, amax, (unsigned int*)qx, n4x);

    const long n4w = (long)N * K / 4;
    k_quant_w<<<(unsigned)((n4w + 255) / 256), 256, 0, stream>>>(
        qweight, (unsigned int*)qw, n4w);

    dim3 grid(N / BN, M / BM);
    k_gemm_fp8<<<grid, 256, 0, stream>>>(qx, qw, amax, w_scale, bias, out, M, N, K);
}
